// VectorQuantizer_33139967656627
// MI455X (gfx1250) — compile-verified, matrix operand order checked by reading, not run on hardware
//
#include <hip/hip_runtime.h>

typedef __bf16 bf16;
typedef __attribute__((ext_vector_type(16))) __bf16 v16bf;
typedef __attribute__((ext_vector_type(8)))  __bf16 v8bf;
typedef __attribute__((ext_vector_type(8)))  float  v8f;
typedef __attribute__((ext_vector_type(4)))  float  v4f;

#define N_VEC   16384          // 16*1024 input vectors
#define N_CODES 8192
#define DIM     512

// output layout (all float)
#define QSTE_OFF  ((size_t)0)
#define QUANT_OFF ((size_t)N_VEC * DIM)                 // 8388608
#define QLOSS_OFF ((size_t)2 * N_VEC * DIM)             // 16777216
#define ELOSS_OFF (QLOSS_OFF + 1)
#define IDX_OFF   (QLOSS_OFF + 2)

// ---------------------------------------------------------------------------
// Kernel 1: f32 -> bf16 conversion + squared-L2 row norms (1 wave per row).
// Rows [0, N_VEC) are inputs, rows [N_VEC, N_VEC+N_CODES) are codebook rows.
// Also zeroes the loss accumulator (row 0, lane 0).
// ---------------------------------------------------------------------------
__global__ __launch_bounds__(256)
void vq_prep(const float* __restrict__ x, const float* __restrict__ e,
             bf16* __restrict__ xbf, bf16* __restrict__ ebf,
             float* __restrict__ xn, float* __restrict__ en,
             float* __restrict__ loss_acc) {
  const int wave = threadIdx.x >> 5;
  const int lane = threadIdx.x & 31;
  const int row  = blockIdx.x * 8 + wave;
  if (row == 0 && lane == 0) *loss_acc = 0.0f;

  const float* src;
  bf16*  dst;
  float* nrm;
  int    r;
  if (row < N_VEC) {
    r = row;          src = x + (size_t)r * DIM; dst = xbf + (size_t)r * DIM; nrm = xn;
  } else {
    r = row - N_VEC;  src = e + (size_t)r * DIM; dst = ebf + (size_t)r * DIM; nrm = en;
  }

  float s = 0.0f;
#pragma unroll
  for (int i = 0; i < DIM / 32; ++i) {
    int d = lane + i * 32;
    float v = src[d];
    s += v * v;
    dst[d] = (bf16)v;
  }
#pragma unroll
  for (int off = 16; off > 0; off >>= 1) s += __shfl_xor(s, off, 32);
  if (lane == 0) nrm[r] = s;
}

// Build one 16-bit WMMA fragment (16 bf16 / lane) from two contiguous
// 16-byte chunks, per the ISA VGPR layout for 16x32 (K pattern
// {half*8+0..7, 16+half*8+0..7}).
__device__ __forceinline__ v16bf load_frag(const bf16* __restrict__ p) {
  v8bf lo = *(const v8bf*)(p);
  v8bf hi = *(const v8bf*)(p + 16);
  v16bf a;
#pragma unroll
  for (int i = 0; i < 8; ++i) { a[i] = lo[i]; a[i + 8] = hi[i]; }
  return a;
}

// ---------------------------------------------------------------------------
// Kernel 2: per-wave argmin over all codes via bf16 WMMA (codes on M axis,
// vectors on N axis). A-tiles are software-pipelined in two 8-slice register
// buffers so loads for one half-tile are in flight while the other half's
// WMMAs execute. One wave (32 threads) handles 16 input vectors.
// ---------------------------------------------------------------------------
__global__ __launch_bounds__(32)
void vq_argmin(const bf16* __restrict__ xbf, const bf16* __restrict__ ebf,
               const float* __restrict__ xn, const float* __restrict__ en,
               const float* __restrict__ x,  const float* __restrict__ e,
               float* __restrict__ out, float* __restrict__ loss_acc) {
  const int lane = threadIdx.x;
  const int n0   = blockIdx.x * 16;
  const int col  = lane & 15;   // vector (B column) / code row owned by lane
  const int half = lane >> 4;   // which 16-lane half

  // ---- B fragments: 16 vectors x full D=512 contraction, VGPR-resident.
  v16bf bfrag[16];
  {
    const bf16* bp = xbf + (size_t)(n0 + col) * DIM + half * 8;
#pragma unroll
    for (int s = 0; s < 16; ++s) bfrag[s] = load_frag(bp + s * 32);
  }
  const float myxn = xn[n0 + col];

  float bestv = 3.4e38f;
  int   besti = 0;

  // Per-lane A row base for code tile 0; advance by 16 rows (16KB) per tile.
  const bf16* ap = ebf + (size_t)col * DIM + half * 8;
  const size_t tile_stride = (size_t)16 * DIM;   // elements

  v16bf bufA[8], bufB[8];
  // Prologue: first half of tile 0.
#pragma unroll
  for (int s = 0; s < 8; ++s) bufA[s] = load_frag(ap + s * 32);

  for (int kc = 0; kc < N_CODES; kc += 16) {
    // Issue loads for second half of the current tile (slices 8..15).
#pragma unroll
    for (int s = 0; s < 8; ++s) bufB[s] = load_frag(ap + 256 + s * 32);

    // Warm the tile after next while we compute.
    __builtin_prefetch(ap + 2 * tile_stride, 0, 1);

    // WMMAs on first half (bufA already resident). Two accumulators break
    // the D->C chain for better XDL pipelining.
    v8f acc0 = {}, acc1 = {};
#pragma unroll
    for (int s = 0; s < 8; s += 2) {
      acc0 = __builtin_amdgcn_wmma_f32_16x16x32_bf16(
          false, bufA[s],     false, bfrag[s],     (short)0, acc0, false, false);
      acc1 = __builtin_amdgcn_wmma_f32_16x16x32_bf16(
          false, bufA[s + 1], false, bfrag[s + 1], (short)0, acc1, false, false);
    }

    // Issue loads for the NEXT tile's first half while bufB WMMAs run.
    const bf16* apn = ap + tile_stride;
    const bf16* apl = (kc + 16 < N_CODES) ? apn : ap;  // stay in bounds
#pragma unroll
    for (int s = 0; s < 8; ++s) bufA[s] = load_frag(apl + s * 32);

    // WMMAs on second half.
#pragma unroll
    for (int s = 0; s < 8; s += 2) {
      acc0 = __builtin_amdgcn_wmma_f32_16x16x32_bf16(
          false, bufB[s],     false, bfrag[8 + s],     (short)0, acc0, false, false);
      acc1 = __builtin_amdgcn_wmma_f32_16x16x32_bf16(
          false, bufB[s + 1], false, bfrag[8 + s + 1], (short)0, acc1, false, false);
    }

    // Distance epilogue: this lane owns codes kc + half*8 + r.
    v4f en0 = *(const v4f*)(en + kc + half * 8);
    v4f en1 = *(const v4f*)(en + kc + half * 8 + 4);
#pragma unroll
    for (int r = 0; r < 8; ++r) {
      float enr = (r < 4) ? en0[r] : en1[r - 4];
      float dot = acc0[r] + acc1[r];
      float dst = myxn + enr - 2.0f * dot;
      int   code = kc + half * 8 + r;
      if (dst < bestv) { bestv = dst; besti = code; }
    }

    ap = apn;
  }

  // Combine the two half-lane partial argmins (ties -> lower index).
  {
    float ov = __shfl_xor(bestv, 16, 32);
    int   oi = __shfl_xor(besti, 16, 32);
    if (ov < bestv || (ov == bestv && oi < besti)) { bestv = ov; besti = oi; }
  }
  if (lane < 16) out[IDX_OFF + n0 + lane] = (float)besti;

  // Gather winning codebook rows in f32 (float4-vectorized), write both
  // output copies, accumulate squared error for the loss scalars.
  float lsum = 0.0f;
#pragma unroll 1
  for (int v = 0; v < 16; ++v) {
    int bi = __shfl(besti, v, 32);
    const v4f* erow = (const v4f*)(e + (size_t)bi * DIM);
    const v4f* xrow = (const v4f*)(x + (size_t)(n0 + v) * DIM);
    v4f* o1 = (v4f*)(out + QSTE_OFF  + (size_t)(n0 + v) * DIM);
    v4f* o2 = (v4f*)(out + QUANT_OFF + (size_t)(n0 + v) * DIM);
#pragma unroll
    for (int i = 0; i < 4; ++i) {
      int idx = lane + i * 32;           // 128 float4 per row
      v4f ev = erow[idx];
      v4f xv = xrow[idx];
      o1[idx] = ev;
      o2[idx] = ev;
#pragma unroll
      for (int c = 0; c < 4; ++c) { float df = ev[c] - xv[c]; lsum += df * df; }
    }
  }
#pragma unroll
  for (int off = 16; off > 0; off >>= 1) lsum += __shfl_xor(lsum, off, 32);
  if (lane == 0) atomicAdd(loss_acc, lsum);
}

// ---------------------------------------------------------------------------
// Kernel 3: finalize the two (numerically identical) loss scalars.
// ---------------------------------------------------------------------------
__global__ void vq_finalize(const float* __restrict__ loss_acc,
                            float* __restrict__ out) {
  float m = *loss_acc * (1.0f / (float)((size_t)N_VEC * DIM));
  out[QLOSS_OFF] = m;
  out[ELOSS_OFF] = m;
}

extern "C" void kernel_launch(void* const* d_in, const int* in_sizes, int n_in,
                              void* d_out, int out_size, void* d_ws, size_t ws_size,
                              hipStream_t stream) {
  const float* x = (const float*)d_in[0];   // (16,1024,512) f32
  const float* e = (const float*)d_in[1];   // (8192,512)    f32
  float* out = (float*)d_out;

  // workspace layout
  char* ws = (char*)d_ws;
  bf16*  xbf  = (bf16*)ws;                                             // 16 MB
  bf16*  ebf  = (bf16*)(ws + (size_t)N_VEC * DIM * 2);                 //  8 MB
  float* xn   = (float*)(ws + (size_t)N_VEC * DIM * 2
                            + (size_t)N_CODES * DIM * 2);              // 64 KB
  float* en   = xn + N_VEC;                                            // 32 KB
  float* loss = en + N_CODES;                                          //  4 B

  // 1) convert + norms (+ zero loss accumulator). 8 waves/block, 1 row/wave.
  vq_prep<<<(N_VEC + N_CODES) / 8, 256, 0, stream>>>(x, e, xbf, ebf, xn, en, loss);

  // 2) WMMA distance + argmin + gather. One wave per 16 vectors.
  vq_argmin<<<N_VEC / 16, 32, 0, stream>>>(xbf, ebf, xn, en, x, e, out, loss);

  // 3) loss scalars.
  vq_finalize<<<1, 1, 0, stream>>>(loss, out);
}
